// SARoutingBlock_35459249996444
// MI455X (gfx1250) — matrix operand order repaired and checked by reading, not compiled, not run on hardware
//
#include <hip/hip_runtime.h>
#include <stdint.h>

// ---------------- problem dims ----------------
constexpr int Bc = 4, Vc = 8, Lc = 256, Cc = 1024, Hc = 16, Oc = 4, DhC = 64;
constexpr int Mrows = Bc * Vc * Lc;          // 8192
constexpr float NEGV = -1e9f;

typedef __attribute__((ext_vector_type(16))) __bf16   v16bf;
typedef __attribute__((ext_vector_type(8)))  float    v8f;
typedef __attribute__((ext_vector_type(8)))  uint32_t v8u;
typedef __attribute__((ext_vector_type(4)))  uint32_t u32x4;
typedef __attribute__((ext_vector_type(8)))  int      i32x8;
typedef __attribute__((ext_vector_type(4)))  int      i32x4;
typedef unsigned short ushort_t;
#define AS3 __attribute__((address_space(3)))

// float -> bf16 round-to-nearest-even
__device__ __forceinline__ ushort_t f2bf(float f) {
    uint32_t u = __builtin_bit_cast(uint32_t, f);
    u += 0x7FFFu + ((u >> 16) & 1u);
    return (ushort_t)(u >> 16);
}

// --------------------------------------------------------------------------
// TDM: 2D bf16 tile (tile_d0 elems/row, tile_d1 rows, row stride in elems)
// global -> LDS with 1-DWORD pad after each 32-DWORD (64 bf16) row, i.e.
// LDS row stride 66 ushorts. D# layout per CDNA5 ISA §8 (group0 + group1).
// Toolchain here is the 6-arg builtin (clang-23 / therock-10.0 headers).
// --------------------------------------------------------------------------
__device__ __forceinline__ void tdm_load_2d_bf16(
    uint32_t lds_off, const void* gptr, uint32_t tile_d0, uint32_t tile_d1,
    uint32_t stride_elems)
{
    uint64_t ga = (uint64_t)(uintptr_t)gptr;
    u32x4 g0 = { 1u,                                  // count=1, is_restore=0
                 lds_off,                             // lds_addr (bytes)
                 (uint32_t)ga,                        // global_addr[31:0]
                 (uint32_t)((ga >> 32) & 0x1FFFFFFu)  // global_addr[56:32]
                     | (2u << 30) };                  // type=2 ("image")
    i32x8 g1;
    g1[0] = (int)((1u << 16)        // data_size = 2 bytes
                | (1u << 20)        // pad_enable
                | (4u << 22)        // pad_interval: 32 DWORDs
                | (0u << 25));      // pad_amount: 1 DWORD
    g1[1] = (int)(tile_d0 << 16);             // tensor_dim0[15:0]
    g1[2] = (int)((tile_d1 & 0xFFFFu) << 16); // tensor_dim1[15:0]
    g1[3] = (int)(tile_d0 << 16);             // tile_dim0
    g1[4] = (int)(tile_d1 & 0xFFFFu);         // tile_dim1 (tile_dim2 = 0)
    g1[5] = (int)stride_elems;                // tensor_dim0_stride[31:0]
    g1[6] = 0;
    g1[7] = 0;
    i32x4 z4 = { 0, 0, 0, 0 };                // groups 2/3 unused (2D tensor)
    i32x8 z8 = { 0, 0, 0, 0, 0, 0, 0, 0 };    // extra descriptor words unused
    __builtin_amdgcn_tensor_load_to_lds(g0, g1, z4, z4, z8, 0);
}

// ==========================================================================
// Kernel 1: routing — avg-pool over L, MLP (ReLU), log_softmax + gumbel,
// softmax -> alphas[b*V+v][O]. One block per (b,v).
// ==========================================================================
__global__ __launch_bounds__(256) void routing_kernel(
    const float* __restrict__ v, const float* __restrict__ W1,
    const float* __restrict__ W2, const float* __restrict__ b2,
    const float* __restrict__ tau, const float* __restrict__ gumbel,
    float* __restrict__ alphas)
{
    __shared__ float pooled[Cc];
    __shared__ float hs[Cc / 2];
    __shared__ float lg[Oc];
    const int bv = blockIdx.x, t = threadIdx.x;
    const float* vb = v + (size_t)bv * Lc * Cc;

    for (int c = t; c < Cc; c += 256) {
        float s = 0.f;
        for (int l = 0; l < Lc; ++l) s += vb[l * Cc + c];
        pooled[c] = s * (1.0f / Lc);
    }
    __syncthreads();
    for (int j = t; j < Cc / 2; j += 256) {
        float s = 0.f;
        for (int c = 0; c < Cc; ++c) s += pooled[c] * W1[c * (Cc / 2) + j];
        hs[j] = s > 0.f ? s : 0.f;
    }
    __syncthreads();
    if (t < Oc) {
        float s = b2[t];
        for (int j = 0; j < Cc / 2; ++j) s += hs[j] * W2[j * Oc + t];
        lg[t] = s;
    }
    __syncthreads();
    if (t == 0) {
        float mx = lg[0];
        for (int o = 1; o < Oc; ++o) mx = fmaxf(mx, lg[o]);
        float se = 0.f;
        for (int o = 0; o < Oc; ++o) se += __expf(lg[o] - mx);
        float lse = mx + __logf(se);
        float it = 1.0f / tau[0];
        float y[Oc], my = -3.0e38f;
        for (int o = 0; o < Oc; ++o) {
            y[o] = (lg[o] - lse + gumbel[bv * Oc + o]) * it;
            my = fmaxf(my, y[o]);
        }
        float s2 = 0.f;
        for (int o = 0; o < Oc; ++o) { y[o] = __expf(y[o] - my); s2 += y[o]; }
        float inv = 1.0f / s2;
        for (int o = 0; o < Oc; ++o) alphas[bv * Oc + o] = y[o] * inv;
    }
}

// ==========================================================================
// Kernel 2/4: WMMA GEMM  C[M,N] = A[M,K] * W[K,N] + bias
//   A: f32 (A_F32) or bf16; C: f32 (OUT_F32) or bf16.
//   Tile 128x128x32, 256 threads = 8 waves, wave grid 4x2, each wave
//   owns 2x4 16x16 accumulators -> 8 v_wmma_f32_16x16x32_bf16 per K-step.
//   Next-tile global_prefetch keeps HBM supply ahead of the WMMA stream.
// ==========================================================================
template <bool A_F32, bool OUT_F32>
__global__ __launch_bounds__(256) void gemm_wmma(
    const void* __restrict__ Aptr, const float* __restrict__ W,
    const float* __restrict__ bias, void* __restrict__ Cptr,
    int M, int N, int K)
{
    constexpr int BM = 128, BN = 128, BK = 32, LDT = 34;
    __shared__ ushort_t As[BM * LDT];   // [m][k], bf16
    __shared__ ushort_t Ws[BN * LDT];   // [n][k], bf16 (transposed)

    const int bm0 = blockIdx.y * BM, bn0 = blockIdx.x * BN;
    const int tid = threadIdx.x, lane = tid & 31, wave = tid >> 5;
    const int wm = wave >> 1;           // 0..3 -> M offset wm*32
    const int wn = wave & 1;            // 0..1 -> N offset wn*64
    const int ln = lane & 15, hi = lane >> 4;

    const float*    Af = (const float*)Aptr;
    const ushort_t* Ab = (const ushort_t*)Aptr;

    v8f acc[2][4];
    #pragma unroll
    for (int mi = 0; mi < 2; ++mi)
        #pragma unroll
        for (int ni = 0; ni < 4; ++ni) acc[mi][ni] = (v8f){};

    for (int k0 = 0; k0 < K; k0 += BK) {
        __syncthreads();
        for (int i = tid; i < BM * BK; i += 256) {
            int m = i >> 5, kk = i & 31;
            int gi = (bm0 + m) * K + k0 + kk;
            As[m * LDT + kk] = A_F32 ? f2bf(Af[gi]) : Ab[gi];
        }
        for (int i = tid; i < BN * BK; i += 256) {
            int kk = i >> 7, n = i & 127;
            Ws[n * LDT + kk] = f2bf(W[(k0 + kk) * N + bn0 + n]);
        }
        if (k0 + BK < K) {  // prefetch next K-tile (global_prefetch_b8)
            int m = tid >> 5, kk = tid & 31;
            int gi = (bm0 + m * 8) * K + k0 + BK + kk;
            __builtin_prefetch(A_F32 ? (const void*)&Af[gi]
                                     : (const void*)&Ab[gi], 0, 0);
            int wk = tid >> 7, n = tid & 127;
            __builtin_prefetch(&W[(k0 + BK + wk * 16) * N + bn0 + n], 0, 0);
        }
        __syncthreads();

        v16bf afrag[2];
        #pragma unroll
        for (int mi = 0; mi < 2; ++mi) {
            v8u au;
            #pragma unroll
            for (int r = 0; r < 8; ++r) {
                int koff = ((r < 4) ? 2 * r : 2 * r + 8) + 8 * hi;
                au[r] = *(const uint32_t*)&As[(wm * 32 + mi * 16 + ln) * LDT + koff];
            }
            afrag[mi] = __builtin_bit_cast(v16bf, au);
        }
        #pragma unroll
        for (int ni = 0; ni < 4; ++ni) {
            v8u bu;
            #pragma unroll
            for (int r = 0; r < 8; ++r) {
                int kk = ((r < 4) ? 2 * r : 2 * r + 8) + 8 * hi;
                bu[r] = *(const uint32_t*)&Ws[(wn * 64 + ni * 16 + ln) * LDT + kk];
            }
            v16bf bfrag = __builtin_bit_cast(v16bf, bu);
            #pragma unroll
            for (int mi = 0; mi < 2; ++mi)
                acc[mi][ni] = __builtin_amdgcn_wmma_f32_16x16x32_bf16(
                    false, afrag[mi], false, bfrag, (short)0, acc[mi][ni],
                    false, false);
        }
    }

    #pragma unroll
    for (int mi = 0; mi < 2; ++mi)
        #pragma unroll
        for (int ni = 0; ni < 4; ++ni)
            #pragma unroll
            for (int r = 0; r < 8; ++r) {
                int m = bm0 + wm * 32 + mi * 16 + hi * 8 + r;
                int n = bn0 + wn * 64 + ni * 16 + ln;
                float val = acc[mi][ni][r] + bias[n];
                if (OUT_F32) ((float*)Cptr)[m * N + n] = val;
                else         ((ushort_t*)Cptr)[m * N + n] = f2bf(val);
            }
}

// ==========================================================================
// Kernel 3: fused routed attention, one block per (b,v,h, 32-row q tile).
//   Q/K tiles staged by the Tensor Data Mover (tensor_load_to_lds with LDS
//   row padding -> stride 66), V transposed manually, S = QK^T/8 via WMMA,
//   per-order masked softmax blended by alphas (wave32 shfl reductions),
//   P(bf16) @ V via WMMA -> bf16 atted.
// LDS: Q 4224 + K 33792 + V^T 33024 + S 32896 + P 16512 + mask 1024 ≈ 119KB.
// ==========================================================================
constexpr int QS_OFF = 0;                         // ushort [32][66]
constexpr int KS_OFF = 4224;                      // ushort [256][66]
constexpr int VT_OFF = 38016;                     // ushort [64][258]  (V^T)
constexpr int SS_OFF = 71040;                     // float  [32][257]
constexpr int PS_OFF = 103936;                    // ushort [32][258]
constexpr int MS_OFF = 120448;                    // uint   [4][64]
constexpr int SMEM_BYTES = 121472;

__global__ __launch_bounds__(256) void attn_kernel(
    const ushort_t* __restrict__ qp, const ushort_t* __restrict__ kp,
    const ushort_t* __restrict__ vp, const unsigned char* __restrict__ masks,
    const float* __restrict__ alphas, ushort_t* __restrict__ attOut)
{
    __shared__ __align__(16) unsigned char smem[SMEM_BYTES];
    ushort_t* Qs  = (ushort_t*)(smem + QS_OFF);
    ushort_t* Ks  = (ushort_t*)(smem + KS_OFF);
    ushort_t* VsT = (ushort_t*)(smem + VT_OFF);
    float*    Ss  = (float*)(smem + SS_OFF);
    ushort_t* Ps  = (ushort_t*)(smem + PS_OFF);
    uint32_t* Mls = (uint32_t*)(smem + MS_OFF);

    const int qt  = blockIdx.x;              // 0..7   (32-row q tile)
    const int bvh = blockIdx.y;              // 0..511
    const int h   = bvh & (Hc - 1);
    const int bv  = bvh >> 4;                // b*V+v
    const int tid = threadIdx.x, lane = tid & 31, wave = tid >> 5;
    const int ln = lane & 15, hi = lane >> 4;
    const int rowBase = bv * Lc;

    // ---- Phase A: stage tiles ----
    const uint32_t lds_base =
        (uint32_t)(uintptr_t)(AS3 unsigned char*)&smem[0];
    if (wave == 0) {
        // Q tile: 32 rows x 64 bf16, K tile: 256 rows x 64 bf16 (TDM DMA)
        tdm_load_2d_bf16(lds_base + QS_OFF,
                         qp + (size_t)(rowBase + qt * 32) * Cc + h * 64,
                         64, 32, Cc);
        tdm_load_2d_bf16(lds_base + KS_OFF,
                         kp + (size_t)rowBase * Cc + h * 64,
                         64, 256, Cc);
        __builtin_amdgcn_s_wait_tensorcnt(0);
    }
    for (int i = tid; i < 256 * 64; i += 256) {  // V transposed (manual)
        int key = i >> 6, d = i & 63;
        VsT[d * 258 + key] = vp[(rowBase + key) * Cc + h * 64 + d];
    }
    {   // mask bytes for this (b,v): 4 orders x 256 keys
        int o = tid >> 6, j = tid & 63;
        Mls[o * 64 + j] =
            *(const uint32_t*)(masks + ((size_t)o * (Bc * Vc) + bv) * Lc + j * 4);
    }
    __syncthreads();

    // ---- Phase B: S = Q K^T (2 M-tiles x 16 N-tiles, 4 per wave) ----
    {
        const int mt = wave & 1;
        const int ntb = (wave >> 1) * 4;
        v8f sacc[4];
        #pragma unroll
        for (int n = 0; n < 4; ++n) sacc[n] = (v8f){};
        #pragma unroll
        for (int ks = 0; ks < 2; ++ks) {
            v8u au;
            #pragma unroll
            for (int r = 0; r < 8; ++r) {
                int koff = ((r < 4) ? 2 * r : 2 * r + 8) + 8 * hi + ks * 32;
                au[r] = *(const uint32_t*)&Qs[(mt * 16 + ln) * 66 + koff];
            }
            v16bf aq = __builtin_bit_cast(v16bf, au);
            #pragma unroll
            for (int nt = 0; nt < 4; ++nt) {
                v8u bu;
                #pragma unroll
                for (int r = 0; r < 8; ++r) {
                    int kk = ((r < 4) ? 2 * r : 2 * r + 8) + 8 * hi + ks * 32;
                    bu[r] = *(const uint32_t*)&Ks[((ntb + nt) * 16 + ln) * 66 + kk];
                }
                v16bf bk = __builtin_bit_cast(v16bf, bu);
                sacc[nt] = __builtin_amdgcn_wmma_f32_16x16x32_bf16(
                    false, aq, false, bk, (short)0, sacc[nt], false, false);
            }
        }
        #pragma unroll
        for (int nt = 0; nt < 4; ++nt)
            #pragma unroll
            for (int r = 0; r < 8; ++r)
                Ss[(mt * 16 + hi * 8 + r) * 257 + (ntb + nt) * 16 + ln] =
                    sacc[nt][r] * 0.125f;   // 1/sqrt(Dh)
    }
    __syncthreads();

    // ---- Phase D: per-order masked softmax blended by alphas ----
    {
        float aW[4];
        #pragma unroll
        for (int o = 0; o < 4; ++o) aW[o] = alphas[bv * 4 + o];
        for (int rr = 0; rr < 4; ++rr) {
            int row = wave * 4 + rr;
            float s[8], p[8];
            #pragma unroll
            for (int j = 0; j < 8; ++j) {
                s[j] = Ss[row * 257 + lane * 8 + j];
                p[j] = 0.f;
            }
            #pragma unroll
            for (int o = 0; o < 4; ++o) {
                uint32_t mu0 = Mls[o * 64 + lane * 2];
                uint32_t mu1 = Mls[o * 64 + lane * 2 + 1];
                float val[8];
                float mx = -3.0e38f;
                #pragma unroll
                for (int j = 0; j < 8; ++j) {
                    uint32_t byte = (j < 4) ? ((mu0 >> (8 * j)) & 0xFF)
                                            : ((mu1 >> (8 * (j - 4))) & 0xFF);
                    val[j] = byte ? NEGV : s[j];
                    mx = fmaxf(mx, val[j]);
                }
                #pragma unroll
                for (int off = 16; off > 0; off >>= 1)
                    mx = fmaxf(mx, __shfl_xor(mx, off, 32));
                float e[8], sum = 0.f;
                #pragma unroll
                for (int j = 0; j < 8; ++j) { e[j] = __expf(val[j] - mx); sum += e[j]; }
                #pragma unroll
                for (int off = 16; off > 0; off >>= 1)
                    sum += __shfl_xor(sum, off, 32);
                float scl = aW[o] / sum;
                #pragma unroll
                for (int j = 0; j < 8; ++j) p[j] += e[j] * scl;
            }
            #pragma unroll
            for (int jj = 0; jj < 4; ++jj) {
                uint32_t packed = (uint32_t)f2bf(p[2 * jj]) |
                                  ((uint32_t)f2bf(p[2 * jj + 1]) << 16);
                *(uint32_t*)&Ps[row * 258 + lane * 8 + 2 * jj] = packed;
            }
        }
    }
    __syncthreads();

    // ---- Phase E: out = P @ V  (2x4 tiles, 1 per wave), K = 256 ----
    {
        const int mt = wave & 1;
        const int nt = wave >> 1;          // 0..3 over Dh
        v8f oacc = (v8f){};
        #pragma unroll
        for (int kb = 0; kb < 8; ++kb) {
            v8u au, bu;
            #pragma unroll
            for (int r = 0; r < 8; ++r) {
                int koff = ((r < 4) ? 2 * r : 2 * r + 8) + 8 * hi + kb * 32;
                au[r] = *(const uint32_t*)&Ps[(mt * 16 + ln) * 258 + koff];
                int kk = ((r < 4) ? 2 * r : 2 * r + 8) + 8 * hi + kb * 32;
                bu[r] = *(const uint32_t*)&VsT[(nt * 16 + ln) * 258 + kk];
            }
            v16bf ap = __builtin_bit_cast(v16bf, au);
            v16bf bvf = __builtin_bit_cast(v16bf, bu);
            oacc = __builtin_amdgcn_wmma_f32_16x16x32_bf16(
                false, ap, false, bvf, (short)0, oacc, false, false);
        }
        #pragma unroll
        for (int r = 0; r < 8; ++r) {
            int m = qt * 32 + mt * 16 + hi * 8 + r;
            int col = h * 64 + nt * 16 + ln;
            attOut[(size_t)(rowBase + m) * Cc + col] = f2bf(oacc[r]);
        }
    }
}

// ==========================================================================
// Launch
// ==========================================================================
extern "C" void kernel_launch(void* const* d_in, const int* in_sizes, int n_in,
                              void* d_out, int out_size, void* d_ws, size_t ws_size,
                              hipStream_t stream) {
    (void)in_sizes; (void)n_in; (void)out_size; (void)ws_size;
    const float* v  = (const float*)d_in[0];
    const float* k  = (const float*)d_in[1];
    const float* q  = (const float*)d_in[2];
    const unsigned char* masks = (const unsigned char*)d_in[3];
    const float* tau    = (const float*)d_in[4];
    const float* gumbel = (const float*)d_in[6];
    const float* Wv = (const float*)d_in[7];
    const float* bv = (const float*)d_in[8];
    const float* Wk = (const float*)d_in[9];
    const float* bk = (const float*)d_in[10];
    const float* Wq = (const float*)d_in[11];
    const float* bq = (const float*)d_in[12];
    const float* Wm = (const float*)d_in[13];
    const float* bm = (const float*)d_in[14];
    const float* W1 = (const float*)d_in[15];
    const float* W2 = (const float*)d_in[16];
    const float* b2 = (const float*)d_in[17];

    char* ws = (char*)d_ws;
    float*    alphas = (float*)ws;                                   // 512 B
    ushort_t* qp  = (ushort_t*)(ws + 1024);
    ushort_t* kp  = qp + (size_t)Mrows * Cc;
    ushort_t* vp  = kp + (size_t)Mrows * Cc;
    ushort_t* att = vp + (size_t)Mrows * Cc;

    routing_kernel<<<Bc * Vc, 256, 0, stream>>>(v, W1, W2, b2, tau, gumbel, alphas);

    dim3 gg(Cc / 128, Mrows / 128);
    gemm_wmma<true, false><<<gg, 256, 0, stream>>>(q, Wq, bq, qp, Mrows, Cc, Cc);
    gemm_wmma<true, false><<<gg, 256, 0, stream>>>(k, Wk, bk, kp, Mrows, Cc, Cc);
    gemm_wmma<true, false><<<gg, 256, 0, stream>>>(v, Wv, bv, vp, Mrows, Cc, Cc);

    attn_kernel<<<dim3(Lc / 32, Bc * Vc * Hc), 256, 0, stream>>>(
        qp, kp, vp, masks, alphas, att);

    gemm_wmma<false, true><<<gg, 256, 0, stream>>>(att, Wm, bm, (float*)d_out,
                                                   Mrows, Cc, Cc);
}
